// RWKVRecurrence_59923383714131
// MI455X (gfx1250) — compile-verified, matrix operand order checked by reading, not run on hardware
//
#include <hip/hip_runtime.h>
#include <hip/hip_bf16.h>
#include <stdint.h>

// ---------------------------------------------------------------------------
// Types matching the gfx1250 WMMA / TDM builtin signatures
// ---------------------------------------------------------------------------
typedef __attribute__((ext_vector_type(16))) __bf16   v16bf;
typedef __attribute__((ext_vector_type(8)))  __bf16   v8bf;
typedef __attribute__((ext_vector_type(8)))  float    v8f;
typedef __attribute__((ext_vector_type(4)))  uint32_t u32x4;
typedef __attribute__((ext_vector_type(8)))  int32_t  i32x8;
typedef __attribute__((ext_vector_type(4)))  int32_t  i32x4;

#define BB 4
#define TT 2048
#define DD 1024
#define MT (BB * TT)   // 8192 rows for all GEMMs

// ---------------------------------------------------------------------------
// TDM helper: 2D tile load (bf16 elements) global -> LDS via Tensor Data Mover
// D# layout per CDNA5 ISA
// ---------------------------------------------------------------------------
__device__ __forceinline__ unsigned lds_off_u32(const void* p) {
  // addrspace(3) offset lives in the low 32 bits of the flat pointer
  return (unsigned)(unsigned long long)p;
}

__device__ __forceinline__ void tdm_load_tile_2d(
    unsigned lds_byte_off, const __bf16* gptr,
    unsigned tile_rows, unsigned tile_cols,
    unsigned tensor_rows, unsigned tensor_cols, unsigned row_stride_elems) {
  unsigned long long ga = (unsigned long long)(const void*)gptr;
  u32x4 g0;
  g0[0] = 1u;                                              // count=1 (valid), user mode
  g0[1] = lds_byte_off;                                    // lds_addr (bytes)
  g0[2] = (uint32_t)ga;                                    // global_addr[31:0]
  g0[3] = (uint32_t)((ga >> 32) & 0x1FFFFFFull)            // global_addr[56:32]
        | (2u << 30);                                      // type = 2 ("image")
  i32x8 g1;
  g1[0] = (int)(1u << 16);                                 // data_size=1 (2 bytes/elem)
  g1[1] = (int)((tensor_cols & 0xFFFFu) << 16);            // tensor_dim0[15:0]
  g1[2] = (int)(((tensor_cols >> 16) & 0xFFFFu)            // tensor_dim0[31:16]
        | ((tensor_rows & 0xFFFFu) << 16));                // tensor_dim1[15:0]
  g1[3] = (int)(((tensor_rows >> 16) & 0xFFFFu)            // tensor_dim1[31:16]
        | ((tile_cols & 0xFFFFu) << 16));                  // tile_dim0
  g1[4] = (int)(tile_rows & 0xFFFFu);                      // tile_dim1 (tile_dim2=0)
  g1[5] = (int)row_stride_elems;                           // tensor_dim0_stride[31:0]
  g1[6] = 0;                                               // stride0 hi / stride1 lo
  g1[7] = 0;
  i32x4 z4 = {0, 0, 0, 0};
#if defined(__clang_major__) && (__clang_major__ >= 23)
  i32x8 z8 = {0, 0, 0, 0, 0, 0, 0, 0};
  __builtin_amdgcn_tensor_load_to_lds(g0, g1, z4, z4, z8, 0);
#else
  __builtin_amdgcn_tensor_load_to_lds(g0, g1, z4, z4, 0);
#endif
}

// ---------------------------------------------------------------------------
// bf16 WMMA GEMM: C[M,N] = A[M,K] (row-major bf16) x Bt[N,K] (N-major bf16)
// 128x128 block tile, 8 waves (wave32), each wave 64x32 = 4x2 WMMA tiles.
// K-tile 64 (two 32-K WMMA halves, 128-byte DMA rows), double-buffered LDS
// (64 KB) fed by the Tensor Data Mover; wave 0 drives the DMA.
// MODE: 0 = f32 store, 1 = sigmoid f32, 2 = resid + acc f32, 3 = sqrelu bf16
// ---------------------------------------------------------------------------
template <int MODE>
__global__ __launch_bounds__(256) void gemm_bf16_wmma(
    const __bf16* __restrict__ A, const __bf16* __restrict__ Bt,
    const float* __restrict__ resid, void* __restrict__ outv,
    int M, int N, int K) {
  constexpr int KT = 64;
  __shared__ __align__(128) __bf16 lds[2][2 * 128 * KT];   // [buf][A(128xKT) | B(128xKT)]
  const int tileM = blockIdx.y * 128;
  const int tileN = blockIdx.x * 128;
  const int wave  = (int)(threadIdx.x >> 5);
  const int lane  = (int)(threadIdx.x & 31);
  const int wm    = wave >> 2;   // 0..1 : 64 rows each
  const int wn    = wave & 3;    // 0..3 : 32 cols each
  const int nsteps = K / KT;

  const size_t arow = (size_t)tileM * K;
  const size_t brow = (size_t)tileN * K;

  if (wave == 0) {
    tdm_load_tile_2d(lds_off_u32(&lds[0][0]),        A + arow, 128, KT, (unsigned)M, (unsigned)K, (unsigned)K);
    tdm_load_tile_2d(lds_off_u32(&lds[0][128 * KT]), Bt + brow, 128, KT, (unsigned)N, (unsigned)K, (unsigned)K);
  }

  v8f acc[4][2] = {};
  const int kb = (lane >> 4) << 3;  // 0 or 8 : K sub-block per half-wave
  const int rr = lane & 15;

  for (int kt = 0; kt < nsteps; ++kt) {
    if (wave == 0) {
      if (kt + 1 < nsteps) {
        const int nb = (kt + 1) & 1;
        tdm_load_tile_2d(lds_off_u32(&lds[nb][0]),        A + arow + (size_t)(kt + 1) * KT, 128, KT, (unsigned)M, (unsigned)K, (unsigned)K);
        tdm_load_tile_2d(lds_off_u32(&lds[nb][128 * KT]), Bt + brow + (size_t)(kt + 1) * KT, 128, KT, (unsigned)N, (unsigned)K, (unsigned)K);
        __builtin_amdgcn_s_wait_tensorcnt(2);  // current buffer done (in-order), next 2 in flight
      } else {
        __builtin_amdgcn_s_wait_tensorcnt(0);
      }
    }
    __syncthreads();

    const __bf16* la = &lds[kt & 1][0];
    const __bf16* lb = &lds[kt & 1][128 * KT];

#pragma unroll
    for (int h = 0; h < 2; ++h) {      // two 32-K WMMA steps per LDS buffer
      const int ko = h * 32;
      v16bf af[4], bfr[2];
#pragma unroll
      for (int i = 0; i < 4; ++i) {
        const __bf16* p = la + (size_t)(wm * 64 + i * 16 + rr) * KT + ko + kb;
        v8bf lo = *(const v8bf*)p;
        v8bf hi = *(const v8bf*)(p + 16);
        af[i] = __builtin_shufflevector(lo, hi, 0, 1, 2, 3, 4, 5, 6, 7, 8, 9, 10, 11, 12, 13, 14, 15);
      }
#pragma unroll
      for (int j = 0; j < 2; ++j) {
        const __bf16* p = lb + (size_t)(wn * 32 + j * 16 + rr) * KT + ko + kb;
        v8bf lo = *(const v8bf*)p;
        v8bf hi = *(const v8bf*)(p + 16);
        bfr[j] = __builtin_shufflevector(lo, hi, 0, 1, 2, 3, 4, 5, 6, 7, 8, 9, 10, 11, 12, 13, 14, 15);
      }
#pragma unroll
      for (int i = 0; i < 4; ++i)
#pragma unroll
        for (int j = 0; j < 2; ++j)
          acc[i][j] = __builtin_amdgcn_wmma_f32_16x16x32_bf16(
              false, af[i], false, bfr[j], (short)0, acc[i][j], false, false);
    }
    __syncthreads();
  }

  // Epilogue. C/D layout: VGPR e holds M = base + e + (lane>=16 ? 8 : 0), N = lane&15.
  const int colBase = tileN + wn * 32;
#pragma unroll
  for (int i = 0; i < 4; ++i) {
#pragma unroll
    for (int j = 0; j < 2; ++j) {
      const int col = colBase + j * 16 + (lane & 15);
      const int rowBase = tileM + wm * 64 + i * 16 + ((lane >> 4) << 3);
#pragma unroll
      for (int e = 0; e < 8; ++e) {
        const int row = rowBase + e;
        const float a = acc[i][j][e];
        const size_t o = (size_t)row * N + col;
        if constexpr (MODE == 0) {
          ((float*)outv)[o] = a;
        } else if constexpr (MODE == 1) {
          ((float*)outv)[o] = 1.f / (1.f + __expf(-a));
        } else if constexpr (MODE == 2) {
          ((float*)outv)[o] = resid[o] + a;
        } else {
          const float rl = a > 0.f ? a : 0.f;
          ((__bf16*)outv)[o] = (__bf16)(rl * rl);
        }
      }
    }
  }
}

// ---------------------------------------------------------------------------
// Weight transpose + fp32 -> bf16: in[K,N] row-major -> out[N,K] row-major
// ---------------------------------------------------------------------------
__global__ __launch_bounds__(256) void transpose_to_bf16(
    const float* __restrict__ in, __bf16* __restrict__ out, int K, int N) {
  __shared__ float tile[32][33];
  const int n0 = blockIdx.x * 32;
  const int k0 = blockIdx.y * 32;
  const int tx = (int)(threadIdx.x & 31);
  const int ty = (int)(threadIdx.x >> 5);
  for (int i = ty; i < 32; i += 8)
    tile[i][tx] = in[(size_t)(k0 + i) * N + (n0 + tx)];
  __syncthreads();
  for (int i = ty; i < 32; i += 8)
    out[(size_t)(n0 + i) * K + (k0 + tx)] = (__bf16)tile[tx][i];
}

// ---------------------------------------------------------------------------
// Token-shift mixing: o_j = bf16( x*m_j + shift(x)*(1-m_j) ), up to 3 outputs
// ---------------------------------------------------------------------------
__global__ __launch_bounds__(256) void mix_shift_kernel(
    const float* __restrict__ x,
    const float* __restrict__ m0, const float* __restrict__ m1, const float* __restrict__ m2,
    __bf16* __restrict__ o0, __bf16* __restrict__ o1, __bf16* __restrict__ o2,
    int total) {
  const int idx = blockIdx.x * blockDim.x + threadIdx.x;
  if (idx >= total) return;
  const int d = idx & (DD - 1);
  const int t = (idx / DD) & (TT - 1);
  const float xc = x[idx];
  const float xs = (t == 0) ? 0.f : x[idx - DD];
  { const float m = m0[d]; o0[idx] = (__bf16)(xc * m + xs * (1.f - m)); }
  if (o1) { const float m = m1[d]; o1[idx] = (__bf16)(xc * m + xs * (1.f - m)); }
  if (o2) { const float m = m2[d]; o2[idx] = (__bf16)(xc * m + xs * (1.f - m)); }
}

// ---------------------------------------------------------------------------
// WKV recurrence, fused with r-gating: rwkv_bf = bf16( r * num/den )
// One thread per (b,d) channel; serial over T with prefetch-ahead.
// ---------------------------------------------------------------------------
__global__ __launch_bounds__(256) void wkv_rwkv_kernel(
    const float* __restrict__ k, const float* __restrict__ v,
    const float* __restrict__ r, const float* __restrict__ time_decay,
    __bf16* __restrict__ rwkv) {
  const int idx = blockIdx.x * blockDim.x + threadIdx.x;
  if (idx >= BB * DD) return;
  const int b = idx / DD;
  const int d = idx & (DD - 1);
  const float ew = __expf(-__expf(time_decay[d]));
  const size_t base = (size_t)b * TT * DD + d;
  float num = 0.f, den = 0.f;
  for (int t = 0; t < TT; ++t) {
    const size_t o = base + (size_t)t * DD;
    if (t + 16 < TT) {
      __builtin_prefetch(k + o + (size_t)16 * DD, 0, 0);  // global_prefetch_b8
      __builtin_prefetch(v + o + (size_t)16 * DD, 0, 0);
    }
    const float ek = __expf(k[o]);
    num = ew * num + ek * v[o];
    den = ew * den + ek;
    const float wkv = num / (den + 1e-6f);
    rwkv[o] = (__bf16)(r[o] * wkv);
  }
}

// ---------------------------------------------------------------------------
// Final: out = x1 + r2 * v2   (r2 already sigmoided by GEMM epilogue mode 1)
// ---------------------------------------------------------------------------
__global__ __launch_bounds__(256) void final_gate_add(
    const float* __restrict__ x1, const float* __restrict__ r2,
    const float* __restrict__ v2, float* __restrict__ out, int total) {
  const int idx = blockIdx.x * blockDim.x + threadIdx.x;
  if (idx < total) out[idx] = x1[idx] + r2[idx] * v2[idx];
}

// ---------------------------------------------------------------------------
// Orchestration
// ---------------------------------------------------------------------------
extern "C" void kernel_launch(void* const* d_in, const int* in_sizes, int n_in,
                              void* d_out, int out_size, void* d_ws, size_t ws_size,
                              hipStream_t stream) {
  const float* x   = (const float*)d_in[0];
  const float* td  = (const float*)d_in[1];
  const float* tmk = (const float*)d_in[2];
  const float* tmv = (const float*)d_in[3];
  const float* tmr = (const float*)d_in[4];
  const float* Wk  = (const float*)d_in[5];
  const float* Wv  = (const float*)d_in[6];
  const float* Wr  = (const float*)d_in[7];
  const float* Wo  = (const float*)d_in[8];
  const float* cmk = (const float*)d_in[9];
  const float* cmr = (const float*)d_in[10];
  const float* Wck = (const float*)d_in[11];
  const float* Wcv = (const float*)d_in[12];
  const float* Wcr = (const float*)d_in[13];
  (void)in_sizes; (void)n_in; (void)out_size; (void)ws_size;

  char* ws = (char*)d_ws;
  size_t off = 0;
  auto alloc = [&](size_t bytes) -> void* {
    void* p = ws + off;
    off += (bytes + 255) & ~(size_t)255;
    return p;
  };

  const size_t act = (size_t)MT * DD;          // 8.4M elements
  __bf16* xk   = (__bf16*)alloc(act * 2);
  __bf16* xv   = (__bf16*)alloc(act * 2);
  __bf16* xr   = (__bf16*)alloc(act * 2);
  __bf16* WkT  = (__bf16*)alloc((size_t)DD * DD * 2);
  __bf16* WvT  = (__bf16*)alloc((size_t)DD * DD * 2);
  __bf16* WrT  = (__bf16*)alloc((size_t)DD * DD * 2);
  __bf16* WoT  = (__bf16*)alloc((size_t)DD * DD * 2);
  __bf16* WcrT = (__bf16*)alloc((size_t)DD * DD * 2);
  __bf16* WckT = (__bf16*)alloc((size_t)4 * DD * DD * 2);  // (4D, D)
  __bf16* WcvT = (__bf16*)alloc((size_t)4 * DD * DD * 2);  // (D, 4D)
  float*  kf   = (float*)alloc(act * 4);
  float*  vf   = (float*)alloc(act * 4);
  float*  rf   = (float*)alloc(act * 4);
  __bf16* rwkv = (__bf16*)alloc(act * 2);
  float*  x1   = (float*)alloc(act * 4);
  __bf16* xk2  = (__bf16*)alloc(act * 2);
  __bf16* xr2  = (__bf16*)alloc(act * 2);
  __bf16* kh   = (__bf16*)alloc(act * 4 * 2);
  float*  v2   = (float*)alloc(act * 4);
  float*  r2   = (float*)alloc(act * 4);

  const int total = (int)act;
  const dim3 blk(256);

  // 1) weight conversion/transposition to bf16 N-major
  transpose_to_bf16<<<dim3(DD / 32, DD / 32), blk, 0, stream>>>(Wk, WkT, DD, DD);
  transpose_to_bf16<<<dim3(DD / 32, DD / 32), blk, 0, stream>>>(Wv, WvT, DD, DD);
  transpose_to_bf16<<<dim3(DD / 32, DD / 32), blk, 0, stream>>>(Wr, WrT, DD, DD);
  transpose_to_bf16<<<dim3(DD / 32, DD / 32), blk, 0, stream>>>(Wo, WoT, DD, DD);
  transpose_to_bf16<<<dim3(DD / 32, DD / 32), blk, 0, stream>>>(Wcr, WcrT, DD, DD);
  transpose_to_bf16<<<dim3(4 * DD / 32, DD / 32), blk, 0, stream>>>(Wck, WckT, DD, 4 * DD);
  transpose_to_bf16<<<dim3(DD / 32, 4 * DD / 32), blk, 0, stream>>>(Wcv, WcvT, 4 * DD, DD);

  // 2) time-mix token shift
  mix_shift_kernel<<<(total + 255) / 256, blk, 0, stream>>>(x, tmk, tmv, tmr, xk, xv, xr, total);

  // 3) k, v, r projections
  gemm_bf16_wmma<0><<<dim3(DD / 128, MT / 128), blk, 0, stream>>>(xk, WkT, nullptr, kf, MT, DD, DD);
  gemm_bf16_wmma<0><<<dim3(DD / 128, MT / 128), blk, 0, stream>>>(xv, WvT, nullptr, vf, MT, DD, DD);
  gemm_bf16_wmma<1><<<dim3(DD / 128, MT / 128), blk, 0, stream>>>(xr, WrT, nullptr, rf, MT, DD, DD);

  // 4) WKV recurrence fused with r-gate
  wkv_rwkv_kernel<<<(BB * DD + 255) / 256, blk, 0, stream>>>(kf, vf, rf, td, rwkv);

  // 5) x1 = x + (r*wkv) @ Wo
  gemm_bf16_wmma<2><<<dim3(DD / 128, MT / 128), blk, 0, stream>>>(rwkv, WoT, x, x1, MT, DD, DD);

  // 6) channel-mix token shift
  mix_shift_kernel<<<(total + 255) / 256, blk, 0, stream>>>(x1, cmk, cmr, nullptr, xk2, xr2, nullptr, total);

  // 7) kh = sqrelu(xk2 @ Wck) ; v2 = kh @ Wcv ; r2 = sigmoid(xr2 @ Wcr)
  gemm_bf16_wmma<3><<<dim3(4 * DD / 128, MT / 128), blk, 0, stream>>>(xk2, WckT, nullptr, kh, MT, 4 * DD, DD);
  gemm_bf16_wmma<0><<<dim3(DD / 128, MT / 128), blk, 0, stream>>>(kh, WcvT, nullptr, v2, MT, DD, 4 * DD);
  gemm_bf16_wmma<1><<<dim3(DD / 128, MT / 128), blk, 0, stream>>>(xr2, WcrT, nullptr, r2, MT, DD, DD);

  // 8) out = x1 + r2 * v2
  final_gate_add<<<(total + 255) / 256, blk, 0, stream>>>(x1, r2, v2, (float*)d_out, total);
}
